// RodimusFlowInner_40114994544706
// MI455X (gfx1250) — compile-verified, matrix operand order checked by reading, not run on hardware
//
#include <hip/hip_runtime.h>
#include <hip/hip_bf16.h>
#include <math.h>

// ---------------------------------------------------------------------------
// RodimusFlow inner kernel for MI455X (gfx1250, wave32, WMMA bf16 + TDM)
// B=4 L=2048 D=2048 M=64 R=128 W=4, chunked parallel scan (C=64)
// ---------------------------------------------------------------------------

#define BB 4
#define LL 2048
#define DD 2048
#define MM 64
#define RR 128
#define CH 64          // scan chunk length
#define SCALE_O 0.125f // 1/sqrt(64)

typedef __attribute__((ext_vector_type(16))) __bf16 bf16x16;
typedef __attribute__((ext_vector_type(8)))  __bf16 bf16x8;
typedef __attribute__((ext_vector_type(8)))  float  f32x8;
typedef __attribute__((ext_vector_type(4)))  unsigned int u32x4;
typedef __attribute__((ext_vector_type(8)))  int          i32x8;
typedef __attribute__((ext_vector_type(4)))  int          i32x4;

#if defined(__has_builtin)
#if __has_builtin(__builtin_amdgcn_tensor_load_to_lds) && \
    __has_builtin(__builtin_amdgcn_s_wait_tensorcnt)
#define HAVE_TDM 1
#endif
#endif
#ifndef HAVE_TDM
#define HAVE_TDM 0
#endif

// ---- WMMA helpers ----------------------------------------------------------
// A/B fragment for V_WMMA_F32_16X16X32_BF16 from a row-major [row][k] matrix.
// Lane layout (ISA 7.12.2, 16-bit A 16x32): lane = 16*half + m,
//   VGPR 0..3 hold K = half*8 + {0..7}, VGPR 4..7 hold K = 16 + half*8 + {0..7}
// => two contiguous 16-byte loads per lane. Works for global or LDS pointers.
__device__ __forceinline__ bf16x16 load_frag(const __bf16* base, int ld,
                                             int row0, int k0, int lane) {
  const int m    = lane & 15;
  const int half = lane >> 4;
  const __bf16* p = base + (size_t)(row0 + m) * ld + k0 + (half << 3);
  bf16x8 lo = *(const bf16x8*)p;
  bf16x8 hi = *(const bf16x8*)(p + 16);
  bf16x16 f;
#pragma unroll
  for (int j = 0; j < 8; ++j) { f[j] = lo[j]; f[j + 8] = hi[j]; }
  return f;
}

__device__ __forceinline__ f32x8 wmma_bf16(bf16x16 a, bf16x16 b, f32x8 c) {
  return __builtin_amdgcn_wmma_f32_16x16x32_bf16(
      /*neg_a=*/false, a, /*neg_b=*/false, b,
      /*c_mod=*/(short)0, c, /*reuse_a=*/false, /*reuse_b=*/false);
}

#if HAVE_TDM
// ---- Tensor Data Mover: async 2-D bf16 tile load, global -> LDS -----------
// D# per ISA cdna5 ch.8: group0 = {count=1, lds_addr, global_addr, type=2},
// group1 = {data_size=2B, tensor dims, 64x64 tile, dim0 stride}. Groups 2/3
// zero (2-D tile). Completion tracked with TENSORcnt.
// Builtin arity differs per toolchain: clang>=23 takes 6 args (extra i32x8
// descriptor group), ROCm 7.2 clang-22 takes 5.
__device__ __forceinline__ void tdm_load_tile_bf16(
    unsigned lds_byte_addr, unsigned long long gaddr, unsigned tensor_d0,
    unsigned tensor_d1, unsigned stride0, unsigned tile_d0, unsigned tile_d1) {
  u32x4 g0;
  g0[0] = 1u;                                  // count=1 (valid), user mode
  g0[1] = lds_byte_addr;                       // LDS dest (bytes)
  g0[2] = (unsigned)gaddr;                     // global_addr[31:0]
  g0[3] = ((unsigned)(gaddr >> 32) & 0x01FFFFFFu) | (2u << 30); // [56:32]|type=2
  i32x8 g1;
  g1[0] = (int)(1u << 16);                     // data_size = 1 -> 2 bytes
  g1[1] = (int)((tensor_d0 & 0xFFFFu) << 16);  // tensor_dim0[15:0]
  g1[2] = (int)((tensor_d0 >> 16) | ((tensor_d1 & 0xFFFFu) << 16));
  g1[3] = (int)((tensor_d1 >> 16) | (tile_d0 << 16)); // tile_dim0
  g1[4] = (int)tile_d1;                        // tile_dim1 (tile_dim2 = 0)
  g1[5] = (int)stride0;                        // tensor_dim0_stride[31:0]
  g1[6] = 0;
  g1[7] = 0;
  const i32x4 z4 = {0, 0, 0, 0};
#if defined(__clang_major__) && (__clang_major__ >= 23)
  const i32x8 z8 = {0, 0, 0, 0, 0, 0, 0, 0};
  __builtin_amdgcn_tensor_load_to_lds(g0, g1, z4, z4, z8, 0);
#else
  __builtin_amdgcn_tensor_load_to_lds(g0, g1, z4, z4, 0);
#endif
}
#endif

// ---------------------------------------------------------------------------
// Kernel 0: convert weights to bf16 (once per launch; tiny)
//   wcat  = [in_proj_w ; mem_gate_w]  (256 x 2048)
//   w0b   = ch_gate_w0 (128 x 2048)   [col=r][k=d]
//   w1b   = ch_gate_w1 (2048 x 128)   [col=d][k=r]
// ---------------------------------------------------------------------------
__global__ __launch_bounds__(256) void cvt_weights_kernel(
    const float* __restrict__ inproj, const float* __restrict__ memw,
    const float* __restrict__ w0, const float* __restrict__ w1,
    __bf16* __restrict__ wcat, __bf16* __restrict__ w0b,
    __bf16* __restrict__ w1b) {
  int i = blockIdx.x * 256 + threadIdx.x; // 0 .. 262143
  if (i < 128 * 2048) {
    wcat[i]              = (__bf16)inproj[i];
    wcat[128 * 2048 + i] = (__bf16)memw[i];
    w0b[i]               = (__bf16)w0[i];
    w1b[i]               = (__bf16)w1[i];
  }
}

// ---------------------------------------------------------------------------
// Kernel 1: causal depthwise conv (W=4) + SiLU -> shift (f32 + bf16), x -> bf16
// One thread per 4 consecutive channels (float4 loads along D).
// ---------------------------------------------------------------------------
__global__ __launch_bounds__(256) void conv_silu_kernel(
    const float* __restrict__ x, const float* __restrict__ w,
    const float* __restrict__ bias, float* __restrict__ shift_f,
    __bf16* __restrict__ shift_b, __bf16* __restrict__ x_b) {
  const long gid = (long)blockIdx.x * blockDim.x + threadIdx.x;
  const int  d   = (int)(gid % (DD / 4)) * 4;
  const long bl  = gid / (DD / 4);
  const int  l   = (int)(bl % LL);
  const long b   = bl / LL;

  float y[4] = {0.f, 0.f, 0.f, 0.f};
#pragma unroll
  for (int i = 0; i < 4; ++i) {
    const int src = l - 3 + i;
    if (src >= 0) {
      const float4 xv = *(const float4*)(x + ((b * LL + src) * (long)DD) + d);
      y[0] += xv.x * w[(d + 0) * 4 + i];
      y[1] += xv.y * w[(d + 1) * 4 + i];
      y[2] += xv.z * w[(d + 2) * 4 + i];
      y[3] += xv.w * w[(d + 3) * 4 + i];
    }
  }
  const long base = (b * LL + l) * (long)DD + d;
  const float4 xc = *(const float4*)(x + base);
  const float xs[4] = {xc.x, xc.y, xc.z, xc.w};
#pragma unroll
  for (int j = 0; j < 4; ++j) {
    const float v = y[j] + bias[d + j];
    const float s = v / (1.f + __expf(-v)); // SiLU
    shift_f[base + j] = s;
    shift_b[base + j] = (__bf16)s;
    x_b[base + j]     = (__bf16)xs[j];
  }
}

// ---------------------------------------------------------------------------
// Kernel 2: generic bf16 WMMA GEMM:  Y[row,col] = sum_k A[row][k] * W[col][k]
// One wave computes a 16-row x 64-col strip (A fragment reused 4x).
// MODE 0: store f32   MODE 1: store bf16   MODE 2: sigmoid(.+bias)*x -> bf16
// ---------------------------------------------------------------------------
template <int MODE>
__global__ __launch_bounds__(256) void gemm_bf16_kernel(
    const __bf16* __restrict__ A, const __bf16* __restrict__ W, int K, int rows,
    int cols, float* __restrict__ Yf, __bf16* __restrict__ Yb,
    const float* __restrict__ bias, const float* __restrict__ xf) {
  const int lane      = threadIdx.x & 31;
  const int waveId    = blockIdx.x * (blockDim.x >> 5) + (threadIdx.x >> 5);
  const int colGroups = cols >> 6;
  const int rt        = waveId / colGroups;
  const int cg        = waveId % colGroups;
  if (rt * 16 >= rows) return; // uniform per wave: EXEC stays all-ones

  f32x8 acc[4] = {};
  for (int k0 = 0; k0 < K; k0 += 32) {
    if (k0 + 64 < K) { // prefetch 2 k-steps ahead (-> global_prefetch_b8)
      __builtin_prefetch(A + (size_t)(rt * 16 + (lane & 15)) * K + k0 + 64, 0, 0);
      __builtin_prefetch(W + (size_t)(cg * 64 + lane) * K + k0 + 64, 0, 0);
      __builtin_prefetch(W + (size_t)(cg * 64 + 32 + lane) * K + k0 + 64, 0, 0);
    }
    const bf16x16 a = load_frag(A, K, rt * 16, k0, lane);
#pragma unroll
    for (int j = 0; j < 4; ++j) {
      const bf16x16 bm = load_frag(W, K, cg * 64 + j * 16, k0, lane);
      acc[j] = wmma_bf16(a, bm, acc[j]);
    }
  }
  const int n = lane & 15, half = lane >> 4;
#pragma unroll
  for (int j = 0; j < 4; ++j) {
    const int col = cg * 64 + j * 16 + n;
#pragma unroll
    for (int v = 0; v < 8; ++v) {
      const int   row = rt * 16 + half * 8 + v;
      const float val = acc[j][v];
      if (MODE == 0) {
        Yf[(size_t)row * cols + col] = val;
      } else if (MODE == 1) {
        Yb[(size_t)row * cols + col] = (__bf16)val;
      } else {
        const float ch = 1.f / (1.f + __expf(-(val + bias[col])));
        Yb[(size_t)row * cols + col] =
            (__bf16)(ch * xf[(size_t)row * cols + col]);
      }
    }
  }
}

// ---------------------------------------------------------------------------
// Kernel 3: per-token gate math.
//   rkmg row = [r(64) | k(64) | sel_pre(64) | tau_pre(64)]
//   sel=softplus(sel_pre+b0), tau=sigmoid(tau_pre+b1), it=sel^tau, g=-sel*tau
//   khat = k/max(||k||,eps) * it
// 64 threads per token, 4 tokens per block.
// ---------------------------------------------------------------------------
__global__ __launch_bounds__(256) void token_post_kernel(
    const float* __restrict__ rkmg, const float* __restrict__ mem_b,
    float* __restrict__ rvec, float* __restrict__ kvec,
    float* __restrict__ gvec) {
  __shared__ float red[256];
  const int tid   = threadIdx.x;
  const int token = blockIdx.x * 4 + (tid >> 6);
  const int j     = tid & 63;
  const float* row = rkmg + (size_t)token * 256;

  const float r  = row[j];
  const float k  = row[64 + j];
  const float m0 = row[128 + j] + mem_b[j];
  const float m1 = row[192 + j] + mem_b[64 + j];

  red[tid] = k * k;
  __syncthreads();
  for (int s = 32; s > 0; s >>= 1) {
    if (j < s) red[tid] += red[tid + s];
    __syncthreads();
  }
  const float nrm = fmaxf(sqrtf(red[tid & ~63]), 1e-12f);

  const float sel = (m0 > 20.f) ? m0 : log1pf(__expf(m0)); // softplus
  const float tau = 1.f / (1.f + __expf(-m1));             // sigmoid
  const float it  = __expf(tau * __logf(sel));             // sel^tau
  const size_t o  = (size_t)token * 64 + j;
  rvec[o] = r;
  kvec[o] = (k / nrm) * it;
  gvec[o] = -sel * tau;
}

// ---------------------------------------------------------------------------
// Kernel 4: chunked parallel scan. grid = (D/64, B); 256 threads (8 waves).
// State tile S_T[d][m] (64x64 f32) lives in LDS for the whole chain.
// Per chunk (all matmuls are 16x16x32 bf16 WMMA):
//   TDM async-loads the 64x64 u-tile into LDS (Pm staging) while threads do
//   the decay prefix sums; then:
//   P  = rh @ khm^T  (causal mask)           rh[t][m], khm[s][m]
//   O  = SCALE*(rh @ Sbf + P @ U) + residual Sbf[d][m], UT[d][t]
//   S  = aend[m] * (S + khT @ U)             khT[m][t]
// ---------------------------------------------------------------------------
__global__ __launch_bounds__(256) void recur_kernel(
    const float* __restrict__ rvec, const float* __restrict__ kvec,
    const float* __restrict__ gvec, const __bf16* __restrict__ u,
    const float* __restrict__ shift, const float* __restrict__ resw,
    float* __restrict__ out) {
  const int d0   = blockIdx.x * 64; // D tile
  const int b    = blockIdx.y;
  const int tid  = threadIdx.x;
  const int lane = tid & 31;
  const int wave = tid >> 5;

  __shared__ float  S[64 * 64];   // state, [d][m]
  __shared__ float  aend[64];
  __shared__ __bf16 rh[64 * 64];  // r*A,   [t][m]
  __shared__ __bf16 khm[64 * 64]; // k/A,   [s][m]  (reused as Sbf [d][m])
  __shared__ __bf16 khT[64 * 64]; // k/A,   [m][t]
  __shared__ __bf16 UT[64 * 72];  // u^T,   [d][t]  (padded rows)
  __shared__ __bf16 Pm[64 * 64];  // TDM staging [t][d], then masked P [t][s]

  for (int i = tid; i < 64 * 64; i += 256) S[i] = 0.f;
  __syncthreads();

  const long tok_base = (long)b * LL;
#if HAVE_TDM
  const unsigned lds_stage = (unsigned)(size_t)(void*)&Pm[0];
#endif

  for (int chunk = 0; chunk < LL / CH; ++chunk) {
    const int tok0 = chunk * CH;

    // ---- phase 1: async U-tile DMA + decay prefix sums --------------------
#if HAVE_TDM
    if (wave == 0) {
      const unsigned long long ga =
          (unsigned long long)(size_t)u +
          ((unsigned long long)(tok_base + tok0) * DD + d0) * 2ull;
      tdm_load_tile_bf16(lds_stage, ga, DD, (unsigned)(BB * LL), DD, 64, 64);
    }
#endif
    if (tid < 64) { // thread owns m; serial prefix over t
      const int m = tid;
      float c = 0.f;
#pragma unroll 4
      for (int t = 0; t < CH; ++t) {
        const long idx = (tok_base + tok0 + t) * 64 + m;
        c += gvec[idx];
        const float ep = __expf(c);
        const float en = __expf(-c);
        rh[t * 64 + m] = (__bf16)(rvec[idx] * ep);
        const float kh = kvec[idx] * en;
        khm[t * 64 + m] = (__bf16)kh;
        khT[m * 64 + t] = (__bf16)kh;
      }
      aend[m] = __expf(c);
    } else {
#if HAVE_TDM
      if (chunk + 1 < LL / CH) { // prefetch next chunk's r/k/g cachelines
        const long nb = (tok_base + tok0 + CH) * 64;
        for (int ln = tid - 64; ln < 128; ln += 192) {
          __builtin_prefetch(gvec + nb + ln * 32, 0, 0);
          __builtin_prefetch(rvec + nb + ln * 32, 0, 0);
          __builtin_prefetch(kvec + nb + ln * 32, 0, 0);
        }
      }
#else
      for (int i = tid - 64; i < 64 * 64; i += 192) { // manual staging
        const int t = i >> 6, d = i & 63;
        UT[d * 72 + t] = u[(tok_base + tok0 + t) * (long)DD + d0 + d];
      }
#endif
    }
#if HAVE_TDM
    __builtin_amdgcn_s_wait_tensorcnt(0); // wave0's DMA done (others: no-op)
    __syncthreads();
    // LDS->LDS transpose of staged tile: Pm[t][d] -> UT[d][t]
    for (int i = tid; i < 64 * 64; i += 256)
      UT[(i & 63) * 72 + (i >> 6)] = Pm[i];
#endif
    __syncthreads();

    // ---- phase 2: P = rh @ khm^T, causal mask, -> bf16 --------------------
#pragma unroll
    for (int tw = 0; tw < 2; ++tw) {
      const int tile = wave * 2 + tw;
      const int rt = tile >> 2, ct = tile & 3;
      f32x8 acc = {};
      for (int k0 = 0; k0 < 64; k0 += 32)
        acc = wmma_bf16(load_frag(rh, 64, rt * 16, k0, lane),
                        load_frag(khm, 64, ct * 16, k0, lane), acc);
      const int n = lane & 15, half = lane >> 4;
      const int s = ct * 16 + n;
#pragma unroll
      for (int v = 0; v < 8; ++v) {
        const int t = rt * 16 + half * 8 + v;
        Pm[t * 64 + s] = (__bf16)((s <= t) ? acc[v] : 0.f);
      }
    }
    __syncthreads();

    // ---- phase 3: Sbf = bf16(S) into khm buffer ---------------------------
    for (int i = tid; i < 64 * 64; i += 256) khm[i] = (__bf16)S[i];
    __syncthreads();

    // ---- phase 4: O = SCALE*(rh@Sbf + Pm@U) + residual --------------------
#pragma unroll
    for (int tw = 0; tw < 2; ++tw) {
      const int tile = wave * 2 + tw;
      const int rt = tile >> 2, ct = tile & 3;
      f32x8 acc = {};
      for (int k0 = 0; k0 < 64; k0 += 32) {
        acc = wmma_bf16(load_frag(rh, 64, rt * 16, k0, lane),
                        load_frag(khm, 64, ct * 16, k0, lane), acc); // rh@S0
        acc = wmma_bf16(load_frag(Pm, 64, rt * 16, k0, lane),
                        load_frag(UT, 72, ct * 16, k0, lane), acc);  // P@U
      }
      const int n = lane & 15, half = lane >> 4;
      const int d = d0 + ct * 16 + n;
      const float rw = resw[d];
#pragma unroll
      for (int v = 0; v < 8; ++v) {
        const int  t   = rt * 16 + half * 8 + v;
        const long gix = (tok_base + tok0 + t) * (long)DD + d;
        out[gix] = SCALE_O * acc[v] + shift[gix] * rw;
      }
    }
    __syncthreads();

    // ---- phase 5: Z = khT @ U ; S = aend*(S + Z) --------------------------
#pragma unroll
    for (int tw = 0; tw < 2; ++tw) {
      const int tile = wave * 2 + tw;
      const int mt = tile >> 2, ct = tile & 3;
      f32x8 acc = {};
      for (int k0 = 0; k0 < 64; k0 += 32)
        acc = wmma_bf16(load_frag(khT, 64, mt * 16, k0, lane),
                        load_frag(UT, 72, ct * 16, k0, lane), acc);
      const int n = lane & 15, half = lane >> 4;
      const int d = ct * 16 + n;
#pragma unroll
      for (int v = 0; v < 8; ++v) {
        const int m = mt * 16 + half * 8 + v;
        S[d * 64 + m] = aend[m] * (S[d * 64 + m] + acc[v]);
      }
    }
    __syncthreads();
  }
}

// ---------------------------------------------------------------------------
extern "C" void kernel_launch(void* const* d_in, const int* in_sizes, int n_in,
                              void* d_out, int out_size, void* d_ws,
                              size_t ws_size, hipStream_t stream) {
  (void)in_sizes; (void)n_in; (void)out_size; (void)ws_size;
  const float* x        = (const float*)d_in[0]; // (B,L,D)
  const float* conv_w   = (const float*)d_in[1]; // (D,4)
  const float* conv_b   = (const float*)d_in[2]; // (D,)
  const float* in_projw = (const float*)d_in[3]; // (128,D)
  const float* w0       = (const float*)d_in[4]; // (128,D)
  const float* w1       = (const float*)d_in[5]; // (D,128)
  const float* b1       = (const float*)d_in[6]; // (D,)
  const float* mem_w    = (const float*)d_in[7]; // (128,D)
  const float* mem_b    = (const float*)d_in[8]; // (128,)
  const float* resw     = (const float*)d_in[9]; // (D,)
  float*       out      = (float*)d_out;

  char*  ws  = (char*)d_ws;
  size_t off = 0;
  auto alloc = [&](size_t bytes) -> void* {
    void* p = ws + off;
    off = (off + bytes + 255) & ~(size_t)255;
    return p;
  };
  const size_t BLD = (size_t)BB * LL * DD;
  const size_t BL  = (size_t)BB * LL;

  float*  shift_f = (float*)alloc(BLD * 4);
  __bf16* shift_b = (__bf16*)alloc(BLD * 2);
  __bf16* x_b     = (__bf16*)alloc(BLD * 2);
  __bf16* u_b     = (__bf16*)alloc(BLD * 2);
  float*  rkmg    = (float*)alloc(BL * 256 * 4);
  __bf16* t_b     = (__bf16*)alloc(BL * 128 * 2);
  float*  rvec    = (float*)alloc(BL * 64 * 4);
  float*  kvec    = (float*)alloc(BL * 64 * 4);
  float*  gvec    = (float*)alloc(BL * 64 * 4);
  __bf16* wcat    = (__bf16*)alloc(256 * 2048 * 2);
  __bf16* w0b     = (__bf16*)alloc(128 * 2048 * 2);
  __bf16* w1b     = (__bf16*)alloc(2048 * 128 * 2);

  // 0) weight conversion
  cvt_weights_kernel<<<1024, 256, 0, stream>>>(in_projw, mem_w, w0, w1, wcat,
                                               w0b, w1b);
  // 1) conv + SiLU
  conv_silu_kernel<<<(int)(BLD / 4 / 256), 256, 0, stream>>>(
      x, conv_w, conv_b, shift_f, shift_b, x_b);
  // 2) rk|mg projection: (8192 x 2048) @ (2048 x 256)  -> rkmg f32
  gemm_bf16_kernel<0><<<256, 256, 0, stream>>>(shift_b, wcat, DD, (int)BL, 256,
                                               rkmg, nullptr, nullptr, nullptr);
  // 3) channel gate stage 1: x @ w0^T -> t (bf16)
  gemm_bf16_kernel<1><<<128, 256, 0, stream>>>(x_b, w0b, DD, (int)BL, RR,
                                               nullptr, t_b, nullptr, nullptr);
  // 4) channel gate stage 2: u = sigmoid(t @ w1^T + b1) * x (bf16)
  gemm_bf16_kernel<2><<<2048, 256, 0, stream>>>(t_b, w1b, RR, (int)BL, DD,
                                                nullptr, u_b, b1, x);
  // 5) per-token gate math -> r, khat, g
  token_post_kernel<<<2048, 256, 0, stream>>>(rkmg, mem_b, rvec, kvec, gvec);
  // 6) chunked scan + residual -> out
  recur_kernel<<<dim3(DD / 64, BB), 256, 0, stream>>>(rvec, kvec, gvec, u_b,
                                                      shift_f, resw, out);
}
